// LongformerModel_24215025615391
// MI455X (gfx1250) — compile-verified, matrix operand order checked by reading, not run on hardware
//
#include <hip/hip_runtime.h>

typedef __attribute__((ext_vector_type(16))) __bf16 bf16x16;
typedef __attribute__((ext_vector_type(8)))  __bf16 bf16x8;
typedef __attribute__((ext_vector_type(8)))  float  f32x8;

#define DMODEL 768
#define NHEAD  12
#define DHEAD  64
#define NLAYER 12
#define DFF    3072
#define SEQ    1024
#define BATCH  4
#define NTOK   (BATCH*SEQ)
#define WIN    256

// Load a 16x32 bf16 A/B-style WMMA fragment from LDS.
// Per ISA 7.12.2 (16-bit A 16x32): lane l holds row (l&15); its 16 elements
// cover K = {K0..K0+7} and {16+K0..16+K0+7} with K0 = 8*(l>=16).
__device__ __forceinline__ bf16x16 lds_frag(const __bf16* base, int K0) {
  bf16x8 lo = *(const bf16x8*)(base + K0);
  bf16x8 hi = *(const bf16x8*)(base + 16 + K0);
  bf16x16 r;
#pragma unroll
  for (int i = 0; i < 8; ++i) { r[i] = lo[i]; r[i + 8] = hi[i]; }
  return r;
}

// ---------------------------------------------------------------------------
// pos = cumsum(mask)*mask + 1  (PAD_IDX = 1)
// ---------------------------------------------------------------------------
__global__ void pos_kernel(const int* __restrict__ mask, int* __restrict__ pos) {
  int b = blockIdx.x;
  int run = 0;
  for (int s = 0; s < SEQ; ++s) {
    int m = mask[b * SEQ + s];
    run += m;
    pos[b * SEQ + s] = run * m + 1;
  }
}

// ---------------------------------------------------------------------------
// x[t][d] = word_emb[ids[t]][d] + pos_emb[pos[t]][d] + type_emb[d]
// ---------------------------------------------------------------------------
__global__ __launch_bounds__(256) void embed_kernel(
    const int* __restrict__ ids, const int* __restrict__ pos,
    const float* __restrict__ wemb, const float* __restrict__ pemb,
    const float* __restrict__ temb, float* __restrict__ x) {
  int t = blockIdx.x;
  const float* w  = wemb + (size_t)ids[t] * DMODEL;
  const float* pe = pemb + (size_t)pos[t] * DMODEL;
  for (int d = threadIdx.x; d < DMODEL; d += 256)
    x[(size_t)t * DMODEL + d] = w[d] + pe[d] + temb[d];
}

// ---------------------------------------------------------------------------
// xout = LayerNorm(xin (+ addv)) * gamma + beta, block per token (D=768)
// ---------------------------------------------------------------------------
__global__ __launch_bounds__(256) void add_ln_kernel(
    const float* __restrict__ xin, const float* __restrict__ addv,
    float* __restrict__ xout, const float* __restrict__ g,
    const float* __restrict__ bta) {
  __shared__ float red[8];
  int t = blockIdx.x;
  int tid = threadIdx.x, lane = tid & 31, wave = tid >> 5;
  float vals[3];
  float sum = 0.f;
#pragma unroll
  for (int i = 0; i < 3; ++i) {
    int d = tid + i * 256;
    float v = xin[(size_t)t * DMODEL + d];
    if (addv) v += addv[(size_t)t * DMODEL + d];
    vals[i] = v;
    sum += v;
  }
#pragma unroll
  for (int off = 16; off; off >>= 1) sum += __shfl_xor(sum, off);
  if (lane == 0) red[wave] = sum;
  __syncthreads();
  float tot = 0.f;
#pragma unroll
  for (int i = 0; i < 8; ++i) tot += red[i];
  float mean = tot * (1.f / DMODEL);
  __syncthreads();
  float vs = 0.f;
#pragma unroll
  for (int i = 0; i < 3; ++i) { float dvi = vals[i] - mean; vs += dvi * dvi; }
#pragma unroll
  for (int off = 16; off; off >>= 1) vs += __shfl_xor(vs, off);
  if (lane == 0) red[wave] = vs;
  __syncthreads();
  float vtot = 0.f;
#pragma unroll
  for (int i = 0; i < 8; ++i) vtot += red[i];
  float inv = rsqrtf(vtot * (1.f / DMODEL) + 1e-5f);
#pragma unroll
  for (int i = 0; i < 3; ++i) {
    int d = tid + i * 256;
    xout[(size_t)t * DMODEL + d] = (vals[i] - mean) * inv * g[d] + bta[d];
  }
}

// ---------------------------------------------------------------------------
// C[M,N] = act(A[M,K] @ Bw[K,N] + bias)   act: 0=none, 1=exact GELU
// 128x128 block tile, 4 waves in a 2x2 grid, wave tile 64x64 (4x4 WMMA
// tiles -> 16 v_wmma per K-step per wave), double-buffered bf16 LDS tiles,
// global_prefetch for tile k+2, fp32 accumulate.
// M%128==0, N%128==0, K%32==0 (all shapes here satisfy this).
// ---------------------------------------------------------------------------
__global__ __launch_bounds__(128) void gemm_bias_act_kernel(
    const float* __restrict__ A, const float* __restrict__ Bw,
    const float* __restrict__ bias, float* __restrict__ C,
    int M, int N, int K, int act) {
  __shared__ __align__(16) __bf16 sA[2][128][40];   // [buf][m][k]
  __shared__ __align__(16) __bf16 sB[2][128][40];   // [buf][n][k] (transposed)
  const int tid  = threadIdx.x;                     // 0..127
  const int lane = tid & 31;
  const int wave = tid >> 5;                        // 0..3
  const int wm = wave >> 1, wn = wave & 1;
  const int l15 = lane & 15;
  const int hf  = (lane >> 4) & 1;
  const int K0  = hf * 8;
  const int bm = blockIdx.y * 128;
  const int bn = blockIdx.x * 128;

  const f32x8 zero8 = {0.f, 0.f, 0.f, 0.f, 0.f, 0.f, 0.f, 0.f};
  f32x8 acc[4][4];
#pragma unroll
  for (int mi = 0; mi < 4; ++mi)
#pragma unroll
    for (int ni = 0; ni < 4; ++ni) acc[mi][ni] = zero8;

  // Cooperative stage of one 128x32 A tile and 32x128 B tile (as [n][k]).
  auto stage = [&](int buf, int k0) {
#pragma unroll
    for (int it = 0; it < 8; ++it) {
      int idx = tid + it * 128;                 // 0..1023 float4 units
      {
        int row = idx >> 3;                     // 0..127
        int c4  = (idx & 7) << 2;               // 0..28
        const float4 v = *(const float4*)(A + (size_t)(bm + row) * K + k0 + c4);
        sA[buf][row][c4 + 0] = (__bf16)v.x; sA[buf][row][c4 + 1] = (__bf16)v.y;
        sA[buf][row][c4 + 2] = (__bf16)v.z; sA[buf][row][c4 + 3] = (__bf16)v.w;
      }
      {
        int row = idx >> 5;                     // k 0..31
        int c4  = (idx & 31) << 2;              // n 0..124
        const float4 v = *(const float4*)(Bw + (size_t)(k0 + row) * N + bn + c4);
        sB[buf][c4 + 0][row] = (__bf16)v.x; sB[buf][c4 + 1][row] = (__bf16)v.y;
        sB[buf][c4 + 2][row] = (__bf16)v.z; sB[buf][c4 + 3][row] = (__bf16)v.w;
      }
    }
  };

  const int nk = K >> 5;
  stage(0, 0);
  __syncthreads();

  for (int ki = 0; ki < nk; ++ki) {
    const int cur = ki & 1;
    if (ki + 1 < nk) stage(cur ^ 1, (ki + 1) << 5);  // fill inactive buffer
    if (ki + 2 < nk) {                                // warm L2 for tile k+2
      int pk = (ki + 2) << 5;
      __builtin_prefetch(A + (size_t)(bm + tid) * K + pk, 0, 0);
      __builtin_prefetch(Bw + (size_t)(pk + (tid >> 2)) * N + bn + (tid & 3) * 32, 0, 0);
    }
    bf16x16 af[4], bfr[4];
#pragma unroll
    for (int mi = 0; mi < 4; ++mi)
      af[mi]  = lds_frag(&sA[cur][wm * 64 + mi * 16 + l15][0], K0);
#pragma unroll
    for (int ni = 0; ni < 4; ++ni)
      bfr[ni] = lds_frag(&sB[cur][wn * 64 + ni * 16 + l15][0], K0);
#pragma unroll
    for (int mi = 0; mi < 4; ++mi)
#pragma unroll
      for (int ni = 0; ni < 4; ++ni)
        acc[mi][ni] = __builtin_amdgcn_wmma_f32_16x16x32_bf16(
            false, af[mi], false, bfr[ni], (short)0, acc[mi][ni], false, false);
    __syncthreads();
  }

  // C/D layout: element r of the v8f holds row (r + 8*hf), col (lane&15).
#pragma unroll
  for (int mi = 0; mi < 4; ++mi)
#pragma unroll
    for (int ni = 0; ni < 4; ++ni) {
      int col = bn + wn * 64 + ni * 16 + l15;
      float bv = bias[col];
#pragma unroll
      for (int r = 0; r < 8; ++r) {
        int row = bm + wm * 64 + mi * 16 + r + hf * 8;
        float v = acc[mi][ni][r] + bv;
        if (act) v = 0.5f * v * (1.f + erff(v * 0.70710678118654752f));
        C[(size_t)row * N + col] = v;
      }
    }
}

// ---------------------------------------------------------------------------
// Sliding-window attention, one block per (chunk, head, batch).
// Streaming (flash-style) softmax over the 768-key window in 32-key tiles.
// q/k/v layout: [b*S + s][h*64 + d] fp32.
// ---------------------------------------------------------------------------
__global__ __launch_bounds__(256) void attn_kernel(
    const float* __restrict__ q, const float* __restrict__ k,
    const float* __restrict__ v, const int* __restrict__ mask,
    float* __restrict__ out) {
  const int chunk = blockIdx.x;   // 0..3
  const int h     = blockIdx.y;   // 0..11
  const int b     = blockIdx.z;   // 0..3
  __shared__ __align__(16) __bf16 ksh[32][72];    // [key][d]
  __shared__ __align__(16) __bf16 vtsh[64][40];   // [d][key] (transposed)
  __shared__ __align__(16) __bf16 psh[8][16][40]; // per-wave P scratch [m][k]
  __shared__ float mskt[32];

  const int tid = threadIdx.x, lane = tid & 31, wave = tid >> 5;
  const int l15 = lane & 15;
  const int hf  = (lane >> 4) & 1;
  const int K0  = hf * 8;
  const int cbase = chunk * WIN;
  const size_t qoff = (size_t)b * SEQ * DMODEL + h * DHEAD;

  const f32x8 zero8 = {0.f, 0.f, 0.f, 0.f, 0.f, 0.f, 0.f, 0.f};
  bf16x16 qa[2][2];
  f32x8 accf[2][4];
  float mstat[2][8], lstat[2][8];
#pragma unroll
  for (int t = 0; t < 2; ++t) {
    int mt = wave * 2 + t;
    int qrow = cbase + mt * 16 + l15;
    const float* qp = q + qoff + (size_t)qrow * DMODEL;
#pragma unroll
    for (int kf = 0; kf < 2; ++kf) {
      int d0 = kf * 32;
#pragma unroll
      for (int i = 0; i < 8; ++i) {
        qa[t][kf][i]     = (__bf16)(qp[d0 + K0 + i] * 0.125f);       // 1/sqrt(64)
        qa[t][kf][i + 8] = (__bf16)(qp[d0 + 16 + K0 + i] * 0.125f);
      }
    }
#pragma unroll
    for (int dt = 0; dt < 4; ++dt) accf[t][dt] = zero8;
#pragma unroll
    for (int r = 0; r < 8; ++r) { mstat[t][r] = -1e30f; lstat[t][r] = 0.f; }
  }

  for (int jj = 0; jj < 3 * WIN; jj += 32) {
    // --- stage K tile (32 keys x 64 d) and V tile transposed, plus mask ----
#pragma unroll
    for (int it = 0; it < 2; ++it) {
      int idx = tid + it * 256;                 // 0..511 (float4 units)
      int key = idx >> 4;
      int d4  = (idx & 15) << 2;
      int kp  = cbase - WIN + jj + key;
      float4 kv = {0.f, 0.f, 0.f, 0.f}, vv = {0.f, 0.f, 0.f, 0.f};
      if (kp >= 0 && kp < SEQ) {
        kv = *(const float4*)(k + qoff + (size_t)kp * DMODEL + d4);
        vv = *(const float4*)(v + qoff + (size_t)kp * DMODEL + d4);
      }
      ksh[key][d4 + 0] = (__bf16)kv.x; ksh[key][d4 + 1] = (__bf16)kv.y;
      ksh[key][d4 + 2] = (__bf16)kv.z; ksh[key][d4 + 3] = (__bf16)kv.w;
      vtsh[d4 + 0][key] = (__bf16)vv.x; vtsh[d4 + 1][key] = (__bf16)vv.y;
      vtsh[d4 + 2][key] = (__bf16)vv.z; vtsh[d4 + 3][key] = (__bf16)vv.w;
    }
    if (tid < 32) {
      int kp = cbase - WIN + jj + tid;
      mskt[tid] = (kp >= 0 && kp < SEQ) ? (float)mask[b * SEQ + kp] : 0.f;
    }
    __syncthreads();

    // B-fragment of K^T == A-layout load of K rows.
    bf16x16 kb[2][2];
#pragma unroll
    for (int sub = 0; sub < 2; ++sub)
#pragma unroll
      for (int kf = 0; kf < 2; ++kf)
        kb[sub][kf] = lds_frag(&ksh[sub * 16 + l15][kf * 32], K0);

#pragma unroll
    for (int t = 0; t < 2; ++t) {
      int mt = wave * 2 + t;
      int mb_ = mt * 16;
      if (jj + 31 < mb_ || jj > mb_ + 15 + 2 * WIN) continue;  // whole tile out of band

      f32x8 s0 = __builtin_amdgcn_wmma_f32_16x16x32_bf16(
          false, qa[t][1], false, kb[0][1], (short)0, zero8, false, false);
      s0 = __builtin_amdgcn_wmma_f32_16x16x32_bf16(
          false, qa[t][0], false, kb[0][0], (short)0, s0, false, false);
      f32x8 s1 = __builtin_amdgcn_wmma_f32_16x16x32_bf16(
          false, qa[t][1], false, kb[1][1], (short)0, zero8, false, false);
      s1 = __builtin_amdgcn_wmma_f32_16x16x32_bf16(
          false, qa[t][0], false, kb[1][0], (short)0, s1, false, false);

      float m0 = mskt[l15], m1 = mskt[16 + l15];
#pragma unroll
      for (int r = 0; r < 8; ++r) {
        int m  = r + hf * 8;
        int iq = mb_ + m;
        int r0 = jj + l15 - WIN - iq;
        int r1 = jj + 16 + l15 - WIN - iq;
        float v0 = (r0 >= -WIN && r0 <= WIN && m0 > 0.f) ? s0[r] : -1e9f;
        float v1 = (r1 >= -WIN && r1 <= WIN && m1 > 0.f) ? s1[r] : -1e9f;
        float tmax = fmaxf(v0, v1);
#pragma unroll
        for (int off = 8; off; off >>= 1) tmax = fmaxf(tmax, __shfl_xor(tmax, off));
        float mold = mstat[t][r];
        float mnew = fmaxf(mold, tmax);
        float sc = __expf(mold - mnew);
        float p0 = __expf(v0 - mnew);
        float p1 = __expf(v1 - mnew);
        float ps = p0 + p1;
#pragma unroll
        for (int off = 8; off; off >>= 1) ps += __shfl_xor(ps, off);
        lstat[t][r] = lstat[t][r] * sc + ps;
        mstat[t][r] = mnew;
#pragma unroll
        for (int dt = 0; dt < 4; ++dt) accf[t][dt][r] *= sc;
        psh[wave][m][l15]      = (__bf16)p0;
        psh[wave][m][16 + l15] = (__bf16)p1;
      }
      asm volatile("" ::: "memory");  // order per-wave LDS store->load (DS in order)

      bf16x16 pa = lds_frag(&psh[wave][l15][0], K0);
#pragma unroll
      for (int dt = 0; dt < 4; ++dt) {
        bf16x16 vb = lds_frag(&vtsh[dt * 16 + l15][0], K0);
        accf[t][dt] = __builtin_amdgcn_wmma_f32_16x16x32_bf16(
            false, pa, false, vb, (short)0, accf[t][dt], false, false);
      }
      asm volatile("" ::: "memory");
    }
    __syncthreads();
  }

#pragma unroll
  for (int t = 0; t < 2; ++t) {
    int mt = wave * 2 + t;
#pragma unroll
    for (int dt = 0; dt < 4; ++dt)
#pragma unroll
      for (int r = 0; r < 8; ++r) {
        int m = r + hf * 8;
        int row = cbase + mt * 16 + m;
        float inv = 1.f / lstat[t][r];
        out[qoff + (size_t)row * DMODEL + dt * 16 + l15] = accf[t][dt][r] * inv;
      }
  }
}

// ---------------------------------------------------------------------------
// out[b] = dot(x[b,0,:768], W_top[0:768]) + fts[b]*W_top[768] + b_top
// ---------------------------------------------------------------------------
__global__ __launch_bounds__(256) void top_kernel(
    const float* __restrict__ x, const float* __restrict__ fts,
    const float* __restrict__ Wt, const float* __restrict__ bt,
    float* __restrict__ out) {
  __shared__ float red[8];
  int b = blockIdx.x, tid = threadIdx.x, lane = tid & 31, wave = tid >> 5;
  float sum = 0.f;
  for (int d = tid; d < DMODEL; d += 256)
    sum += x[(size_t)b * SEQ * DMODEL + d] * Wt[d];
#pragma unroll
  for (int off = 16; off; off >>= 1) sum += __shfl_xor(sum, off);
  if (lane == 0) red[wave] = sum;
  __syncthreads();
  if (tid == 0) {
    float tot = 0.f;
    for (int i = 0; i < 8; ++i) tot += red[i];
    out[b] = tot + fts[b] * Wt[DMODEL] + bt[0];
  }
}

// ---------------------------------------------------------------------------
extern "C" void kernel_launch(void* const* d_in, const int* in_sizes, int n_in,
                              void* d_out, int out_size, void* d_ws, size_t ws_size,
                              hipStream_t stream) {
  (void)in_sizes; (void)n_in; (void)out_size; (void)ws_size;
  const int*   ids     = (const int*)  d_in[0];
  const int*   mask    = (const int*)  d_in[1];
  const float* fts     = (const float*)d_in[2];
  const float* wemb    = (const float*)d_in[3];
  const float* pemb    = (const float*)d_in[4];
  const float* temb    = (const float*)d_in[5];
  const float* eln_s   = (const float*)d_in[6];
  const float* eln_b   = (const float*)d_in[7];
  const float* Wq      = (const float*)d_in[8];
  const float* bq      = (const float*)d_in[9];
  const float* Wk      = (const float*)d_in[10];
  const float* bk      = (const float*)d_in[11];
  const float* Wv      = (const float*)d_in[12];
  const float* bv      = (const float*)d_in[13];
  const float* Wo      = (const float*)d_in[14];
  const float* bo      = (const float*)d_in[15];
  const float* ln1_s   = (const float*)d_in[16];
  const float* ln1_b   = (const float*)d_in[17];
  const float* Wi      = (const float*)d_in[18];
  const float* bi      = (const float*)d_in[19];
  const float* Wf      = (const float*)d_in[20];
  const float* bf_     = (const float*)d_in[21];
  const float* ln2_s   = (const float*)d_in[22];
  const float* ln2_b   = (const float*)d_in[23];
  const float* Wtop    = (const float*)d_in[24];
  const float* btop    = (const float*)d_in[25];
  float* outp = (float*)d_out;

  char* wsp = (char*)d_ws;
  auto carve = [&](size_t bytes) -> void* {
    void* p = (void*)wsp;
    wsp += (bytes + 255) & ~(size_t)255;
    return p;
  };
  int*   pos = (int*)  carve((size_t)NTOK * 4);
  float* x   = (float*)carve((size_t)NTOK * DMODEL * 4);
  float* qb  = (float*)carve((size_t)NTOK * DMODEL * 4);
  float* kb  = (float*)carve((size_t)NTOK * DMODEL * 4);
  float* vb  = (float*)carve((size_t)NTOK * DMODEL * 4);
  float* ab  = (float*)carve((size_t)NTOK * DMODEL * 4);
  float* ob  = (float*)carve((size_t)NTOK * DMODEL * 4);
  float* hb  = (float*)carve((size_t)NTOK * DFF * 4);

  pos_kernel<<<BATCH, 1, 0, stream>>>(mask, pos);
  embed_kernel<<<NTOK, 256, 0, stream>>>(ids, pos, wemb, pemb, temb, x);
  add_ln_kernel<<<NTOK, 256, 0, stream>>>(x, nullptr, x, eln_s, eln_b);

  const dim3 gProj(DMODEL / 128, NTOK / 128);   // 6 x 32
  const dim3 gFF1(DFF / 128, NTOK / 128);       // 24 x 32
  const dim3 gAttn(SEQ / WIN, NHEAD, BATCH);    // 4 x 12 x 4

  for (int l = 0; l < NLAYER; ++l) {
    const size_t wDD = (size_t)l * DMODEL * DMODEL;
    const size_t wDF = (size_t)l * DMODEL * DFF;
    gemm_bias_act_kernel<<<gProj, 128, 0, stream>>>(x, Wq + wDD, bq + l * DMODEL, qb,
                                                    NTOK, DMODEL, DMODEL, 0);
    gemm_bias_act_kernel<<<gProj, 128, 0, stream>>>(x, Wk + wDD, bk + l * DMODEL, kb,
                                                    NTOK, DMODEL, DMODEL, 0);
    gemm_bias_act_kernel<<<gProj, 128, 0, stream>>>(x, Wv + wDD, bv + l * DMODEL, vb,
                                                    NTOK, DMODEL, DMODEL, 0);
    attn_kernel<<<gAttn, 256, 0, stream>>>(qb, kb, vb, mask, ab);
    gemm_bias_act_kernel<<<gProj, 128, 0, stream>>>(ab, Wo + wDD, bo + l * DMODEL, ob,
                                                    NTOK, DMODEL, DMODEL, 0);
    add_ln_kernel<<<NTOK, 256, 0, stream>>>(x, ob, x, ln1_s + l * DMODEL, ln1_b + l * DMODEL);
    gemm_bias_act_kernel<<<gFF1, 128, 0, stream>>>(x, Wi + wDF, bi + l * DFF, hb,
                                                   NTOK, DFF, DMODEL, 1);
    gemm_bias_act_kernel<<<gProj, 128, 0, stream>>>(hb, Wf + wDF, bf_ + l * DMODEL, ob,
                                                    NTOK, DMODEL, DFF, 0);
    add_ln_kernel<<<NTOK, 256, 0, stream>>>(x, ob, x, ln2_s + l * DMODEL, ln2_b + l * DMODEL);
  }
  top_kernel<<<BATCH, 256, 0, stream>>>(x, fts, Wtop, btop, outp);
}
